// causalAttention_11647951306945
// MI455X (gfx1250) — compile-verified, matrix operand order checked by reading, not run on hardware
//
#include <hip/hip_runtime.h>
#include <hip/hip_bf16.h>

// ---------------------------------------------------------------------------
// Causal attention for MI455X (gfx1250): bf16 WMMA + Tensor Data Mover.
//   N = 8192 tokens, D = 1024 features.
// All GEMMs are C = A * B^T. Workgroup = 256 thr (8 waves) computing a
// 128x128 C tile; A/B 128x32 bf16 tiles staged global->LDS by TDM
// (tensor_load_to_lds, TENSORcnt, double buffered, TDM pad feature gives
// an 80B LDS row stride for bank spreading), fragments read via ds_load_b128,
// math via v_wmma_f32_16x16x32_bf16.
// S/P is a materialized 8192^2 bf16 buffer (128 MB -> fits 192 MB L2).
// ---------------------------------------------------------------------------

#define NT 8192
#define DM 1024
#define LDS_LD 40  // padded LDS row stride in elements (64B data + 16B pad)

typedef __attribute__((ext_vector_type(16))) __bf16 v16bf;
typedef __attribute__((ext_vector_type(8)))  __bf16 v8bf;
typedef __attribute__((ext_vector_type(8)))  float  v8f;
typedef __attribute__((ext_vector_type(4)))  unsigned int u32x4;
typedef __attribute__((ext_vector_type(8)))  int i32x8;
typedef __attribute__((ext_vector_type(4)))  int i32x4;

#if __has_builtin(__builtin_amdgcn_tensor_load_to_lds) && \
    __has_builtin(__builtin_amdgcn_s_wait_tensorcnt)
#define HAVE_TDM 1
#else
#define HAVE_TDM 0
#endif

// Load a 16(row) x 32(K) bf16 tile (row-major, row stride `ld` elements) into
// the CDNA5 WMMA 16-bit A-fragment layout:
//   lane = m + 16*khalf ; elems 0..7 = K[8*khalf..+7], elems 8..15 = K[16+8*khalf..+7]
__device__ inline v16bf load_frag_rm(const __bf16* __restrict__ base, int ld) {
    const int lane = threadIdx.x & 31;
    const int m    = lane & 15;
    const int half = lane >> 4;
    const __bf16* r = base + (size_t)m * ld + 8 * half;
    v8bf lo = *reinterpret_cast<const v8bf*>(r);
    v8bf hi = *reinterpret_cast<const v8bf*>(r + 16);
    v16bf out;
#pragma unroll
    for (int i = 0; i < 8; ++i) { out[i] = lo[i]; out[i + 8] = hi[i]; }
    return out;
}

__device__ inline v8f wmma_bf16(v16bf a, v16bf b, v8f c) {
    return __builtin_amdgcn_wmma_f32_16x16x32_bf16(false, a, false, b, (short)0, c,
                                                   false, false);
}

#if HAVE_TDM
// TDM: async-load a 128(rows) x 32(cols) bf16 tile, row stride `lda` elements,
// into LDS at byte offset lds_off with 4-dword padding every 16 dwords
// (LDS row stride = 80B = LDS_LD elements). Tensor window == tile, so no OOB.
__device__ __attribute__((always_inline)) inline void
tdm_load_tile(unsigned int lds_off, const __bf16* gbase, unsigned int lda) {
    const unsigned long long ga = (unsigned long long)(uintptr_t)gbase;
    u32x4 g0;
    g0[0] = 1u;                                    // count=1 (valid), user mode
    g0[1] = lds_off;                               // lds_addr
    g0[2] = (unsigned int)ga;                      // global_addr[31:0]
    g0[3] = (unsigned int)((ga >> 32) & 0x1FFFFFFu) | (2u << 30);  // addr[56:32], type=2
    i32x8 g1;
    g1[0] = (int)((1u << 16)      // data_size = 1 -> 2 bytes
                | (1u << 20)      // pad_enable
                | (3u << 22)      // pad_interval = 3 -> every 16 dwords (64B row)
                | (3u << 25));    // pad_amount   = 3 -> 4 dwords (16B)
    g1[1] = (int)(32u << 16);     // tensor_dim0[15:0]=32 (bits 63:48)
    g1[2] = (int)(128u << 16);    // tensor_dim0[31:16]=0 | tensor_dim1[15:0]=128
    g1[3] = (int)(32u << 16);     // tensor_dim1[31:16]=0 | tile_dim0=32
    g1[4] = (int)128u;            // tile_dim1=128, tile_dim2=0 (2D)
    g1[5] = (int)lda;             // tensor_dim0_stride[31:0]
    g1[6] = 0;                    // stride[47:32]=0, tensor_dim1_stride lo=0
    g1[7] = 0;
    i32x4 z4 = {0, 0, 0, 0};
    i32x8 z8 = {0, 0, 0, 0, 0, 0, 0, 0};
    // clang-23 / amdgpu-toolchain 6-arg form: (g0, g1, g2, g3, g4, cpol)
    __builtin_amdgcn_tensor_load_to_lds(g0, g1, z4, z4, z8, 0);
}
#endif

// ---------------------------------------------------------------------------
// Shared 128x128 GEMM core: acc[2][4] per wave (wave = 32 rows x 64 cols).
__device__ __attribute__((always_inline)) inline void
gemm_core_128x128(const __bf16* __restrict__ A, int lda, int wgrow,
                  const __bf16* __restrict__ B, int ldb, int wgcol,
                  int kmax, __bf16* As, __bf16* Bs, v8f acc[2][4]) {
    const int wave = threadIdx.x >> 5;
    const int wm = wave >> 1, wn = wave & 1;
    const bool issuer = (wave == 0);
    const int nk = kmax >> 5;
    const unsigned int aoff0 = (unsigned int)(uintptr_t)As;
    const unsigned int boff0 = (unsigned int)(uintptr_t)Bs;
    const unsigned int bufbytes = 128u * LDS_LD * 2u;

#if HAVE_TDM
    if (issuer) {
        tdm_load_tile(aoff0, A + (size_t)wgrow * lda, (unsigned int)lda);
        tdm_load_tile(boff0, B + (size_t)wgcol * ldb, (unsigned int)ldb);
    }
#endif
    for (int t = 0; t < nk; ++t) {
        const int cur = t & 1;
        __bf16* Ac = As + cur * 128 * LDS_LD;
        __bf16* Bc = Bs + cur * 128 * LDS_LD;
#if HAVE_TDM
        if (issuer) {
            if (t + 1 < nk) {  // prefetch next tile pair into other buffer
                tdm_load_tile(aoff0 + (cur ^ 1) * bufbytes,
                              A + (size_t)wgrow * lda + (t + 1) * 32, (unsigned int)lda);
                tdm_load_tile(boff0 + (cur ^ 1) * bufbytes,
                              B + (size_t)wgcol * ldb + (t + 1) * 32, (unsigned int)ldb);
                __builtin_amdgcn_s_wait_tensorcnt(2);  // current pair landed
            } else {
                __builtin_amdgcn_s_wait_tensorcnt(0);
            }
        }
        __syncthreads();  // LDS tile visible to all waves
#else
        __syncthreads();
        for (int e = threadIdx.x; e < 128 * 32; e += 256) {
            const int r = e >> 5, c = e & 31;
            Ac[r * LDS_LD + c] = A[(size_t)(wgrow + r) * lda + t * 32 + c];
            Bc[r * LDS_LD + c] = B[(size_t)(wgcol + r) * ldb + t * 32 + c];
        }
        __syncthreads();
#endif
        v16bf a0 = load_frag_rm(Ac + (wm * 32) * LDS_LD, LDS_LD);
        v16bf a1 = load_frag_rm(Ac + (wm * 32 + 16) * LDS_LD, LDS_LD);
        v16bf b[4];
#pragma unroll
        for (int j = 0; j < 4; ++j)
            b[j] = load_frag_rm(Bc + (wn * 64 + 16 * j) * LDS_LD, LDS_LD);
#pragma unroll
        for (int j = 0; j < 4; ++j) {
            acc[0][j] = wmma_bf16(a0, b[j], acc[0][j]);
            acc[1][j] = wmma_bf16(a1, b[j], acc[1][j]);
        }
#if HAVE_TDM
        __syncthreads();  // reads done before TDM overwrites this buffer
#endif
    }
}

// ---------------------------------------------------------------------------
__global__ void cvt_f32_bf16(const float* __restrict__ src, __bf16* __restrict__ dst,
                             long n) {
    long i = (long)blockIdx.x * blockDim.x + threadIdx.x;
    if (i < n) dst[i] = (__bf16)src[i];
}

// ---------------------------------------------------------------------------
// QKV projection: out[m,n] = sum_k X[m,k] * W[n,k].  grid = (8, 64, 3)
// z==0 -> Q row-major, z==1 -> K row-major, z==2 -> V stored transposed.
__global__ __launch_bounds__(256) void qkv_gemm(const __bf16* __restrict__ Xb,
                                                const __bf16* __restrict__ Wb,
                                                __bf16* __restrict__ Qb,
                                                __bf16* __restrict__ Kb,
                                                __bf16* __restrict__ Vt) {
    __shared__ __align__(16) __bf16 As[2 * 128 * LDS_LD];
    __shared__ __align__(16) __bf16 Bs[2 * 128 * LDS_LD];
    const int which = blockIdx.z;
    const __bf16* W = Wb + (size_t)which * DM * DM;
    const int wave = threadIdx.x >> 5;
    const int wm = wave >> 1, wn = wave & 1;
    const int wgrow = blockIdx.y * 128, wgcol = blockIdx.x * 128;
    const int row0 = wgrow + wm * 32, col0 = wgcol + wn * 64;

    v8f acc[2][4];
#pragma unroll
    for (int i = 0; i < 2; ++i)
#pragma unroll
        for (int j = 0; j < 4; ++j) acc[i][j] = (v8f)(0.0f);

    gemm_core_128x128(Xb, DM, wgrow, W, DM, wgcol, DM, As, Bs, acc);

    const int lane = threadIdx.x & 31;
    const int n = lane & 15, half = lane >> 4;
    if (which < 2) {
        __bf16* O = (which == 0) ? Qb : Kb;
#pragma unroll
        for (int i = 0; i < 2; ++i)
#pragma unroll
            for (int j = 0; j < 4; ++j) {
                const int gi0 = row0 + 16 * i + 8 * half;
                const int gc  = col0 + 16 * j + n;
#pragma unroll
                for (int r = 0; r < 8; ++r)
                    O[(size_t)(gi0 + r) * DM + gc] = (__bf16)acc[i][j][r];
            }
    } else {
#pragma unroll
        for (int i = 0; i < 2; ++i)
#pragma unroll
            for (int j = 0; j < 4; ++j) {
                const int gi0 = row0 + 16 * i + 8 * half;
                const int gc  = col0 + 16 * j + n;
#pragma unroll
                for (int r = 0; r < 8; ++r)
                    Vt[(size_t)gc * NT + (gi0 + r)] = (__bf16)acc[i][j][r];
            }
    }
}

// ---------------------------------------------------------------------------
// S = Q * K^T (raw scores, bf16), causal mask -> -inf. grid = (64 jb, 64 ib).
__global__ __launch_bounds__(256) void qk_gemm(const __bf16* __restrict__ Qb,
                                               const __bf16* __restrict__ Kb,
                                               __bf16* __restrict__ SP) {
    const int jb = blockIdx.x, ib = blockIdx.y;
    if (jb > ib) return;  // fully masked block (uniform exit, no barrier issues)
    __shared__ __align__(16) __bf16 As[2 * 128 * LDS_LD];
    __shared__ __align__(16) __bf16 Bs[2 * 128 * LDS_LD];
    const int wave = threadIdx.x >> 5;
    const int wm = wave >> 1, wn = wave & 1;
    const int wgrow = ib * 128, wgcol = jb * 128;
    const int row0 = wgrow + wm * 32, col0 = wgcol + wn * 64;

    v8f acc[2][4];
#pragma unroll
    for (int i = 0; i < 2; ++i)
#pragma unroll
        for (int j = 0; j < 4; ++j) acc[i][j] = (v8f)(0.0f);

    gemm_core_128x128(Qb, DM, wgrow, Kb, DM, wgcol, DM, As, Bs, acc);

    const int lane = threadIdx.x & 31;
    const int n = lane & 15, half = lane >> 4;
#pragma unroll
    for (int i = 0; i < 2; ++i)
#pragma unroll
        for (int j = 0; j < 4; ++j) {
            const int gi0 = row0 + 16 * i + 8 * half;
            const int gj  = col0 + 16 * j + n;
#pragma unroll
            for (int r = 0; r < 8; ++r) {
                const int gi = gi0 + r;
                SP[(size_t)gi * NT + gj] =
                    (gj <= gi) ? (__bf16)acc[i][j][r] : (__bf16)(-INFINITY);
            }
        }
}

// ---------------------------------------------------------------------------
// Row softmax in place over S (bf16): max, then p = exp((s-m)/32), sum.
__global__ __launch_bounds__(256) void softmax_rows(__bf16* __restrict__ SP,
                                                    float* __restrict__ rowsum) {
    const int i = blockIdx.x;
    const int limit = ((i >> 7) + 1) << 7;  // end of diagonal 128-block
    __bf16* row = SP + (size_t)i * NT;
    const int lane = threadIdx.x & 31;
    const int wave = threadIdx.x >> 5;
    __shared__ float red[8];

    float m = -INFINITY;
    for (int c = threadIdx.x; c <= i; c += 256) m = fmaxf(m, (float)row[c]);
#pragma unroll
    for (int off = 16; off > 0; off >>= 1) m = fmaxf(m, __shfl_xor(m, off, 32));
    if (lane == 0) red[wave] = m;
    __syncthreads();
    m = -INFINITY;
#pragma unroll
    for (int w = 0; w < 8; ++w) m = fmaxf(m, red[w]);
    __syncthreads();

    const float inv_scale = 0.03125f;  // 1/sqrt(1024)
    float s = 0.0f;
    for (int c = threadIdx.x; c < limit; c += 256) {
        float v = (float)row[c];
        float p = __expf((v - m) * inv_scale);  // exp(-inf)==0 for masked cols
        row[c] = (__bf16)p;                     // same-address overwrite: race-free
        s += p;
    }
#pragma unroll
    for (int off = 16; off > 0; off >>= 1) s += __shfl_xor(s, off, 32);
    if (lane == 0) red[wave] = s;
    __syncthreads();
    if (threadIdx.x == 0) {
        float t = 0.0f;
#pragma unroll
        for (int w = 0; w < 8; ++w) t += red[w];
        rowsum[i] = t;
    }
}

// ---------------------------------------------------------------------------
// O = (P * Vt^T) / rowsum.  grid = (8 nb, 64 ib); triangular K loop.
__global__ __launch_bounds__(256) void pv_gemm(const __bf16* __restrict__ SP,
                                               const __bf16* __restrict__ Vt,
                                               const float* __restrict__ rowsum,
                                               float* __restrict__ out) {
    __shared__ __align__(16) __bf16 As[2 * 128 * LDS_LD];
    __shared__ __align__(16) __bf16 Bs[2 * 128 * LDS_LD];
    const int ib = blockIdx.y;
    const int wave = threadIdx.x >> 5;
    const int wm = wave >> 1, wn = wave & 1;
    const int wgrow = ib * 128, wgcol = blockIdx.x * 128;
    const int row0 = wgrow + wm * 32, col0 = wgcol + wn * 64;
    const int kmax = (ib + 1) * 128;

    v8f acc[2][4];
#pragma unroll
    for (int i = 0; i < 2; ++i)
#pragma unroll
        for (int j = 0; j < 4; ++j) acc[i][j] = (v8f)(0.0f);

    gemm_core_128x128(SP, NT, wgrow, Vt, NT, wgcol, kmax, As, Bs, acc);

    const int lane = threadIdx.x & 31;
    const int n = lane & 15, half = lane >> 4;
#pragma unroll
    for (int i = 0; i < 2; ++i)
#pragma unroll
        for (int r = 0; r < 8; ++r) {
            const int gi = row0 + 16 * i + 8 * half + r;
            const float inv = 1.0f / rowsum[gi];
#pragma unroll
            for (int j = 0; j < 4; ++j) {
                const int gc = col0 + 16 * j + n;
                out[(size_t)gi * DM + gc] = acc[i][j][r] * inv;
            }
        }
}

// ---------------------------------------------------------------------------
extern "C" void kernel_launch(void* const* d_in, const int* in_sizes, int n_in,
                              void* d_out, int out_size, void* d_ws, size_t ws_size,
                              hipStream_t stream) {
    const float* x  = (const float*)d_in[0];
    const float* wq = (const float*)d_in[1];
    const float* wk = (const float*)d_in[2];
    const float* wv = (const float*)d_in[3];
    float* out = (float*)d_out;

    char* p = (char*)d_ws;
    auto take = [&](size_t bytes) {
        char* r = p;
        p += (bytes + 255) & ~(size_t)255;
        return r;
    };
    __bf16* Xb = (__bf16*)take((size_t)NT * DM * 2);      // 16 MB
    __bf16* Wb = (__bf16*)take((size_t)3 * DM * DM * 2);  //  6 MB
    __bf16* Qb = (__bf16*)take((size_t)NT * DM * 2);      // 16 MB
    __bf16* Kb = (__bf16*)take((size_t)NT * DM * 2);      // 16 MB
    __bf16* Vt = (__bf16*)take((size_t)DM * NT * 2);      // 16 MB (transposed V)
    __bf16* SP = (__bf16*)take((size_t)NT * NT * 2);      // 128 MB (scores -> probs)
    float* rowsum = (float*)take((size_t)NT * 4);

    const long nx = (long)NT * DM, nw = (long)DM * DM;
    cvt_f32_bf16<<<(nx + 255) / 256, 256, 0, stream>>>(x, Xb, nx);
    cvt_f32_bf16<<<(nw + 255) / 256, 256, 0, stream>>>(wq, Wb + 0 * nw, nw);
    cvt_f32_bf16<<<(nw + 255) / 256, 256, 0, stream>>>(wk, Wb + 1 * nw, nw);
    cvt_f32_bf16<<<(nw + 255) / 256, 256, 0, stream>>>(wv, Wb + 2 * nw, nw);

    qkv_gemm<<<dim3(DM / 128, NT / 128, 3), 256, 0, stream>>>(Xb, Wb, Qb, Kb, Vt);
    qk_gemm<<<dim3(NT / 128, NT / 128), 256, 0, stream>>>(Qb, Kb, SP);
    softmax_rows<<<dim3(NT), 256, 0, stream>>>(SP, rowsum);
    pv_gemm<<<dim3(DM / 128, NT / 128), 256, 0, stream>>>(SP, Vt, rowsum, out);
}
